// ContrastiveLoss_83743272337874
// MI455X (gfx1250) — compile-verified, matrix operand order checked by reading, not run on hardware
//
#include <hip/hip_runtime.h>
#include <hip/hip_bf16.h>
#include <math.h>

// ---------------------------------------------------------------------------
// Contrastive (NT-Xent-style) loss over per-pixel feature sets.
//   B=64, C=256, H=W=32.  Per pixel: 128 vectors of dim 256 ->
//   cosine Gram (WMMA f16) -> masked log-softmax (T=0.5) -> mean of positives.
// ---------------------------------------------------------------------------

typedef __attribute__((ext_vector_type(16))) _Float16 v16h;
typedef __attribute__((ext_vector_type(8)))  _Float16 v8h;
typedef __attribute__((ext_vector_type(8)))  float    v8f;

namespace {
constexpr int kB   = 64;     // batch
constexpr int kC   = 256;    // channels (K dim)
constexpr int kHW  = 1024;   // pixels = grid size
constexpr int kNF  = 128;    // 2B feature vectors per pixel
constexpr int kFS  = 264;    // f16 row stride (256 + 8 pad, keeps 16B align, bank-skewed)
constexpr float kInvT = 2.0f;      // 1 / temperature
constexpr float kEps  = 1.1920929e-7f;  // FLT_EPSILON (jnp.finfo(f32).eps)

// dynamic LDS layout
constexpr unsigned kFeatBytes = kNF * kFS * sizeof(_Float16);  // 67584
constexpr unsigned kSsqOff    = kFeatBytes;                    // 256 floats
constexpr unsigned kInvOff    = kSsqOff + 256 * sizeof(float); // 128 floats
constexpr unsigned kWsumOff   = kInvOff + 128 * sizeof(float); // 8 floats
constexpr unsigned kSmemBytes = kWsumOff + 8 * sizeof(float);  // ~69 KB
}

__global__ void __launch_bounds__(256)
contrastive_pixel_kernel(const float* __restrict__ inp,
                         const float* __restrict__ tgt,
                         float* __restrict__ partial) {
  extern __shared__ __align__(16) char smem[];
  _Float16* feats = reinterpret_cast<_Float16*>(smem);
  float*    ssq   = reinterpret_cast<float*>(smem + kSsqOff);
  float*    inv   = reinterpret_cast<float*>(smem + kInvOff);
  float*    wsum  = reinterpret_cast<float*>(smem + kWsumOff);

  const int t   = threadIdx.x;
  const int pix = blockIdx.x;               // = h*W + w, stride-1 in memory

  // ---------- Phase 1: strided global load, sum-of-squares, f16 pack to LDS -
  {
    const int row = t >> 1;                 // 0..127 (feature vector index)
    const int ch0 = (t & 1) * 128;          // each thread: half a row (128 ch)
    const int b   = row & (kB - 1);
    const float* src  = (row < kB) ? inp : tgt;
    const float* base = src + (size_t)b * kC * kHW + pix;   // + c*kHW per chan
    float ss = 0.0f;
    for (int c8 = 0; c8 < 128; c8 += 8) {
      v8h pk;
#pragma unroll
      for (int u = 0; u < 8; ++u) {
        float v = base[(size_t)(ch0 + c8 + u) * kHW];
        ss += v * v;
        pk[u] = (_Float16)v;
      }
      *reinterpret_cast<v8h*>(&feats[row * kFS + ch0 + c8]) = pk;  // ds_store_b128
    }
    ssq[t] = ss;
  }
  __syncthreads();

  // ---------- Phase 2: per-row inverse norm (eps-clamped) --------------------
  if (t < kNF) {
    float n2 = ssq[2 * t] + ssq[2 * t + 1];
    inv[t] = 1.0f / fmaxf(sqrtf(n2), kEps);
  }
  __syncthreads();

  // ---------- Phase 3: Gram matrix via v_wmma_f32_16x16x32_f16 ---------------
  // Wave wv owns rows [wv*16, wv*16+16) x all 128 cols: acc[8] tiles, K=256.
  const int wv    = t >> 5;
  const int lane  = t & 31;
  const int i0    = wv * 16;
  const int lrow  = lane & 15;
  const int khalf = (lane >> 4) * 16;       // hi half-wave takes K+16 chunk
  const int mrow  = i0 + lrow;

  v8f acc[8];
#pragma unroll
  for (int j = 0; j < 8; ++j) acc[j] = {};

#pragma unroll
  for (int ks = 0; ks < 8; ++ks) {
    const int k0 = ks * 32 + khalf;
    union { v16h v; v8h h[2]; } ua;
    ua.h[0] = *reinterpret_cast<const v8h*>(&feats[mrow * kFS + k0]);
    ua.h[1] = *reinterpret_cast<const v8h*>(&feats[mrow * kFS + k0 + 8]);
#pragma unroll
    for (int j = 0; j < 8; ++j) {
      const int ncol = j * 16 + lrow;       // B tile = F^T slice (Gram: same loads)
      union { v16h v; v8h h[2]; } ub;
      ub.h[0] = *reinterpret_cast<const v8h*>(&feats[ncol * kFS + k0]);
      ub.h[1] = *reinterpret_cast<const v8h*>(&feats[ncol * kFS + k0 + 8]);
      acc[j] = __builtin_amdgcn_wmma_f32_16x16x32_f16(
          /*neg_a=*/false, ua.v, /*neg_b=*/false, ub.v,
          /*c_mod=*/(short)0, acc[j], /*reuse_a=*/false, /*reuse_b=*/false);
    }
  }

  // ---------- Phase 4: masked log-softmax per row, in registers --------------
  // C/D layout: reg r, lane L -> (M = i0 + r + 8*(L>>4), N16 = L&15).
  // Row m lives in one 16-lane half-wave: 8 regs x 16 lanes = 128 cols.
  const int n16 = lrow;                     // column-within-tile for this lane
  float icol[8];
#pragma unroll
  for (int j = 0; j < 8; ++j) icol[j] = inv[j * 16 + n16];

  float lsum = 0.0f;
#pragma unroll
  for (int r = 0; r < 8; ++r) {
    const int m   = i0 + r + ((lane >> 4) * 8);   // global row index
    const float im = inv[m];
    float v[8];
    float mx = -INFINITY;
#pragma unroll
    for (int j = 0; j < 8; ++j) {
      const int col = j * 16 + n16;
      float c = acc[j][r] * im * icol[j];         // cosine similarity
      if (col == m) c = -INFINITY;                // mask self-similarity
      v[j] = c;
      mx = fmaxf(mx, c);
    }
    // reduce max over the 16 lanes holding this row (xor 1,2,4,8 stays in half)
    for (int o = 8; o > 0; o >>= 1) mx = fmaxf(mx, __shfl_xor(mx, o, 32));
    float se = 0.0f;
#pragma unroll
    for (int j = 0; j < 8; ++j) se += __expf((v[j] - mx) * kInvT);
    for (int o = 8; o > 0; o >>= 1) se += __shfl_xor(se, o, 32);

    // positive pair: col = m ^ 64 -> held by lane with n16 == (m & 15),
    // in register j_pos = (m ^ 64) >> 4. Loss contrib = lse - s_pos/T.
    if (n16 == (m & 15)) {
      const float spos = acc[(m ^ 64) >> 4][r] * im * inv[m ^ 64];
      lsum += (kInvT * mx + __logf(se)) - kInvT * spos;
    }
  }
  // full-wave reduce of per-lane contributions
  for (int o = 16; o > 0; o >>= 1) lsum += __shfl_xor(lsum, o, 32);
  if (lane == 0) wsum[wv] = lsum;
  __syncthreads();

  if (t == 0) {
    float tot = 0.0f;
#pragma unroll
    for (int q = 0; q < 8; ++q) tot += wsum[q];
    partial[pix] = tot;                      // sum over this pixel's 128 rows
  }
}

// Deterministic fixed-order final reduction of the 1024 per-pixel partials.
__global__ void __launch_bounds__(256)
contrastive_reduce_kernel(const float* __restrict__ partial,
                          float* __restrict__ out) {
  __shared__ float red[256];
  const int t = threadIdx.x;
  float s = partial[t] + partial[t + 256] + partial[t + 512] + partial[t + 768];
  red[t] = s;
  __syncthreads();
  for (int o = 128; o > 0; o >>= 1) {
    if (t < o) red[t] += red[t + o];
    __syncthreads();
  }
  if (t == 0) out[0] = red[0] * (1.0f / ((float)kHW * (float)kNF));
}

extern "C" void kernel_launch(void* const* d_in, const int* in_sizes, int n_in,
                              void* d_out, int out_size, void* d_ws, size_t ws_size,
                              hipStream_t stream) {
  (void)in_sizes; (void)n_in; (void)out_size; (void)ws_size;
  const float* inp = (const float*)d_in[0];
  const float* tgt = (const float*)d_in[1];
  float* out     = (float*)d_out;
  float* partial = (float*)d_ws;             // 1024 floats of scratch

  contrastive_pixel_kernel<<<kHW, 256, kSmemBytes, stream>>>(inp, tgt, partial);
  contrastive_reduce_kernel<<<1, 256, 0, stream>>>(partial, out);
}
